// DigitConvolutionalModel_86801289052343
// MI455X (gfx1250) — matrix operand order repaired and checked by reading, not compiled, not run on hardware
//
#include <hip/hip_runtime.h>
#include <hip/hip_bf16.h>

typedef __attribute__((ext_vector_type(16))) __bf16 v16bf;
typedef __attribute__((ext_vector_type(8)))  float  v8f;
typedef unsigned int u32x4 __attribute__((ext_vector_type(4)));
typedef int          i32x8 __attribute__((ext_vector_type(8)));
typedef int          i32x4 __attribute__((ext_vector_type(4)));

#define BATCH   65536
#define IMG     28
#define NPIX    784           // 28*28
#define OUTHW   26
#define FLAT    676           // 26*26
#define HID     100
#define NCLS    10
#define KPAD    800           // 784 padded to 25*32
#define NPAD    112           // 100 padded to 7*16
#define K2PAD   128           // 112 padded to 4*32
#define LWS     808           // LDS row stride in bf16 (=404 dwords, bank-conflict-free)

union AFrag { v16bf v; __bf16 e[16]; float4 f4[2]; };
union BFrag { v16bf v; __bf16 e[16]; float4 f4[2]; };

// ---------------------------------------------------------------------------
// Prep 1: fold conv3x3 into W1 -> Weff[n][k] (bf16, n-major k-contiguous,
// zero padded to 112 x 800 so the main GEMM needs no edge handling)
// ---------------------------------------------------------------------------
__global__ void prep_weff(const float* __restrict__ conv_w,
                          const float* __restrict__ w1,
                          __bf16* __restrict__ weff) {
    int idx = blockIdx.x * blockDim.x + threadIdx.x;
    if (idx >= NPAD * KPAD) return;
    int n = idx / KPAD;
    int k = idx % KPAD;
    float v = 0.0f;
    if (n < HID && k < NPIX) {
        int i = k / IMG, j = k % IMG;
        #pragma unroll
        for (int di = 0; di < 3; ++di) {
            #pragma unroll
            for (int dj = 0; dj < 3; ++dj) {
                int pi = i - di, pj = j - dj;
                if (pi >= 0 && pi < OUTHW && pj >= 0 && pj < OUTHW)
                    v += conv_w[di * 3 + dj] * w1[n * FLAT + pi * OUTHW + pj];
            }
        }
    }
    weff[idx] = (__bf16)v;
}

// ---------------------------------------------------------------------------
// Prep 2: W2 (10x100) -> bf16 padded 16 x 128, n-major k-contiguous
// ---------------------------------------------------------------------------
__global__ void prep_w2(const float* __restrict__ w2, __bf16* __restrict__ w2b) {
    int idx = blockIdx.x * blockDim.x + threadIdx.x;
    if (idx >= 16 * K2PAD) return;
    int n = idx / K2PAD;
    int k = idx % K2PAD;
    float v = (n < NCLS && k < HID) ? w2[n * HID + k] : 0.0f;
    w2b[idx] = (__bf16)v;
}

__device__ inline void cvt8(const float4& a, const float4& b, __bf16* e) {
    e[0] = (__bf16)a.x; e[1] = (__bf16)a.y; e[2] = (__bf16)a.z; e[3] = (__bf16)a.w;
    e[4] = (__bf16)b.x; e[5] = (__bf16)b.y; e[6] = (__bf16)b.z; e[7] = (__bf16)b.w;
}

// ---------------------------------------------------------------------------
// Fused:  out = relu(x @ Weff + b1) @ W2b + b2
// 256 threads = 8 wave32; each wave owns a 16-row batch tile.
// Weff staged in LDS by the Tensor Data Mover (iterate mode adds +8 bf16 of
// row padding so B-fragment ds_load_b128 reads are bank-conflict-free).
// ---------------------------------------------------------------------------
__global__ void __launch_bounds__(256)
fused_mlp(const float*  __restrict__ x,
          const __bf16* __restrict__ weff,
          const __bf16* __restrict__ w2b,
          const float*  __restrict__ b1,
          const float*  __restrict__ b2,
          float*        __restrict__ out) {
    __shared__ __align__(16) unsigned char smem[NPAD * LWS * 2];   // 180,992 B
    __bf16* lw = (__bf16*)smem;

    const int tid  = threadIdx.x;
    const int wave = tid >> 5;        // 0..7
    const int lane = tid & 31;
    const int kh   = lane >> 4;       // K-half selector (WMMA 16-bit A/B layout)
    const int nl   = lane & 15;       // M (A) or N (B/C) index within tile
    const int rowBase = blockIdx.x * 128 + wave * 16;

    // ---- TDM: DMA Weff (global, L2-resident) into LDS, one wave issues ----
    if (wave == 0) {
        unsigned long long ga = (unsigned long long)(const void*)weff;
        unsigned int lds_base = (unsigned int)(size_t)smem;   // addr[31:0] = LDS byte offset
        u32x4 g0;
        g0.x = 1u;                                            // count=1, user descriptor
        g0.y = lds_base;                                      // lds_addr
        g0.z = (unsigned int)ga;                              // global_addr[31:0]
        g0.w = (unsigned int)((ga >> 32) & 0x01FFFFFFu) | 0x80000000u;  // addr[56:32] | type=2
        i32x8 g1;
        g1[0] = (2 << 16) | (1 << 19);     // data_size=4B, iterate_enable=1
        g1[1] = (KPAD / 2) << 16;          // tensor_dim0 = 400 dwords (low16)
        g1[2] = (1 << 16);                 // tensor_dim0 hi16=0 | tensor_dim1 = 1
        g1[3] = (KPAD / 2) << 16;          // tensor_dim1 hi16=0 | tile_dim0 = 400
        g1[4] = 1;                         // tile_dim1 = 1, tile_dim2 = 0
        g1[5] = KPAD / 2;                  // tensor_dim0_stride = 400 (low32)
        g1[6] = 0;                         // stride hi16 = 0, dim1_stride lo16 = 0
        g1[7] = 0;
        // iterate mode: 112 rows, global += 400 dwords, lds += 404 dwords
        i32x4 g2 = { 0, LWS / 2, KPAD / 2, (NPAD - 1) << 16 };
        i32x4 g3 = { 0, 0, 0, 0 };
        i32x8 g4 = { 0, 0, 0, 0, 0, 0, 0, 0 };   // extra operand of 6-arg builtin
        __builtin_amdgcn_tensor_load_to_lds(g0, g1, g2, g3, g4, 0);
        __builtin_amdgcn_s_wait_tensorcnt(0);
    }
    __syncthreads();

    const float* xrow = x + (size_t)(rowBase + nl) * NPIX;
    const __bf16* lwbase = lw + (size_t)nl * LWS + kh * 8;

    v8f acc[7];
    #pragma unroll
    for (int t = 0; t < 7; ++t) acc[t] = (v8f){0.f,0.f,0.f,0.f,0.f,0.f,0.f,0.f};

    // ---- GEMM1: (16 x 800) @ (800 x 112), 24 full K-steps + tail ----
    for (int kb = 0; kb < 768; kb += 32) {
        AFrag a;
        const float4* p0 = (const float4*)(xrow + kb + kh * 8);
        float4 q0 = p0[0], q1 = p0[1];
        const float4* p1 = (const float4*)(xrow + kb + 16 + kh * 8);
        float4 q2 = p1[0], q3 = p1[1];
        cvt8(q0, q1, &a.e[0]);
        cvt8(q2, q3, &a.e[8]);
        if (kb + 32 < NPIX)
            __builtin_prefetch(xrow + kb + 32, 0, 1);

        // batch all 7 B-fragments (distinct regs) so ds loads overlap
        BFrag bfr[7];
        #pragma unroll
        for (int t = 0; t < 7; ++t) {
            const float4* wp = (const float4*)(lwbase + kb + (size_t)t * 16 * LWS);
            bfr[t].f4[0] = wp[0];
            bfr[t].f4[1] = wp[2];
        }
        #pragma unroll
        for (int t = 0; t < 7; ++t)
            acc[t] = __builtin_amdgcn_wmma_f32_16x16x32_bf16(
                false, a.v, false, bfr[t].v, (short)0, acc[t], false, false);
    }
    {   // tail K-step kb=768: x cols 768..783 valid, 784..799 are zero in Weff
        const int kb = 768;
        AFrag a;
        const float4* p0 = (const float4*)(xrow + kb + kh * 8);
        float4 q0 = p0[0], q1 = p0[1];
        float4 qz = make_float4(0.f, 0.f, 0.f, 0.f);
        cvt8(q0, q1, &a.e[0]);
        cvt8(qz, qz, &a.e[8]);
        BFrag bfr[7];
        #pragma unroll
        for (int t = 0; t < 7; ++t) {
            const float4* wp = (const float4*)(lwbase + kb + (size_t)t * 16 * LWS);
            bfr[t].f4[0] = wp[0];
            bfr[t].f4[1] = wp[2];
        }
        #pragma unroll
        for (int t = 0; t < 7; ++t)
            acc[t] = __builtin_amdgcn_wmma_f32_16x16x32_bf16(
                false, a.v, false, bfr[t].v, (short)0, acc[t], false, false);
    }

    // ---- all waves done reading Weff; overlay h1 buffer on same LDS ----
    __syncthreads();
    __bf16* hb = (__bf16*)smem;   // hb[(wave*16 + m) * 128 + n2]  (32 KB used)

    #pragma unroll
    for (int t = 0; t < 7; ++t) {
        int hn = t * 16 + nl;
        float bv = b1[hn < HID ? hn : 0];
        float bias1 = (hn < HID) ? bv : 0.0f;
        #pragma unroll
        for (int r = 0; r < 8; ++r) {
            float hv = acc[t][r] + bias1;
            hv = fmaxf(hv, 0.0f);
            hb[(wave * 16 + r + kh * 8) * K2PAD + hn] = (__bf16)hv;
        }
    }
    {   // zero pad columns 112..127 (32 lanes x 8 = 16 rows x 16 cols)
        __bf16* zp = hb + (wave * 16 + nl) * K2PAD + 112 + kh * 8;
        #pragma unroll
        for (int i = 0; i < 8; ++i) zp[i] = (__bf16)0.0f;
    }
    __syncthreads();

    // ---- GEMM2: (16 x 128) @ (128 x 16), 4 K-steps ----
    v8f acc2 = (v8f){0.f,0.f,0.f,0.f,0.f,0.f,0.f,0.f};
    #pragma unroll
    for (int t2 = 0; t2 < 4; ++t2) {
        int kb2 = t2 * 32;
        AFrag a2;
        a2.f4[0] = *(const float4*)(hb + (wave * 16 + nl) * K2PAD + kb2 + kh * 8);
        a2.f4[1] = *(const float4*)(hb + (wave * 16 + nl) * K2PAD + kb2 + 16 + kh * 8);
        BFrag b2f;
        const float4* wp = (const float4*)(w2b + nl * K2PAD + kb2 + kh * 8);
        b2f.f4[0] = wp[0];
        b2f.f4[1] = wp[2];
        acc2 = __builtin_amdgcn_wmma_f32_16x16x32_bf16(
            false, a2.v, false, b2f.v, (short)0, acc2, false, false);
    }

    // ---- + b2, store 10 logits per row ----
    if (nl < NCLS) {
        float bo = b2[nl];
        #pragma unroll
        for (int r = 0; r < 8; ++r) {
            int m = r + kh * 8;
            out[(size_t)(rowBase + m) * NCLS + nl] = acc2[r] + bo;
        }
    }
}

extern "C" void kernel_launch(void* const* d_in, const int* in_sizes, int n_in,
                              void* d_out, int out_size, void* d_ws, size_t ws_size,
                              hipStream_t stream) {
    const float* x      = (const float*)d_in[0];
    const float* conv_w = (const float*)d_in[1];
    const float* w1     = (const float*)d_in[2];
    const float* b1     = (const float*)d_in[3];
    const float* w2     = (const float*)d_in[4];
    const float* b2     = (const float*)d_in[5];
    float* out = (float*)d_out;

    __bf16* weff = (__bf16*)d_ws;                                   // 112*800*2 = 179200 B
    __bf16* w2b  = (__bf16*)((char*)d_ws + NPAD * KPAD * 2);        // 16*128*2  =   4096 B

    prep_weff<<<(NPAD * KPAD + 255) / 256, 256, 0, stream>>>(conv_w, w1, weff);
    prep_w2<<<(16 * K2PAD + 255) / 256, 256, 0, stream>>>(w2, w2b);
    fused_mlp<<<BATCH / 128, 256, 0, stream>>>(x, weff, w2b, b1, b2, out);
}
